// Decoder_28922309771884
// MI455X (gfx1250) — matrix operand order repaired and checked by reading, not compiled, not run on hardware
//
#include <hip/hip_runtime.h>

// ---------------------------------------------------------------------------
// Types for gfx1250 WMMA (wave32). Storage uses ushort for bf16 to avoid any
// scalar __bf16 arithmetic; fragments are bit-cast through a union.
// ---------------------------------------------------------------------------
typedef __attribute__((ext_vector_type(16))) __bf16 v16bf;
typedef __attribute__((ext_vector_type(8)))  float  v8f;

struct __attribute__((aligned(16))) U4 { unsigned int x, y, z, w; };
union Frag16 { v16bf v; U4 u[2]; };

__device__ __forceinline__ unsigned short f2bf(float f) {
    union { float f; unsigned int u; } x; x.f = f;
    unsigned int r = x.u + 0x7FFFu + ((x.u >> 16) & 1u);   // round-to-nearest-even
    return (unsigned short)(r >> 16);
}

// gfx1250 async global->LDS DMA (ASYNCcnt path, bypasses VGPRs).
// LDS byte offset = low 32 bits of the generic pointer (ISA aperture rule).
__device__ __forceinline__ void async_load_b128(const void* gaddr, void* lds_ptr) {
    unsigned lds = (unsigned)(unsigned long long)lds_ptr;
    asm volatile("global_load_async_to_lds_b128 %0, %1, off"
                 :: "v"(lds), "v"(gaddr) : "memory");
}
__device__ __forceinline__ void wait_async0() {
    asm volatile("s_wait_asynccnt 0x0" ::: "memory");
}

static constexpr int cS  = 1024;
static constexpr int cH  = 4096;
static constexpr int cNH = 32;
static constexpr int cHD = 128;
static constexpr int cFF = 11008;
static constexpr float cEPS   = 1e-5f;
static constexpr float cGAMMA = 64.0f;
static constexpr float cSCALE = 0.08838834764831845f;     // 1/sqrt(128)

// Epilogue variants
enum {
    EPI_F32 = 0,       // store f32
    EPI_BF16,          // store bf16
    EPI_SILU_BF16,     // silu(acc) -> bf16
    EPI_SILU_F32,      // silu(acc) -> f32
    EPI_HASH_BF16,     // t=acc*GAMMA; t/(1+|t|) -> bf16
    EPI_MASK_F32,      // acc*scale + causal(-1e9) -> f32
    EPI_ADDRES_F32,    // acc + aux[m,n] -> f32
    EPI_MULAUX_BF16    // acc * aux[m,n] -> bf16
};

// ---------------------------------------------------------------------------
// Tiled WMMA GEMM:  C[M,N] = A[M,K] (bf16, row-major) * Bt[N,K]^T (bf16)
// Block tile 128x128x32, 256 threads (8 waves), wave tile 32x64 (2x4 wmma).
// Double-buffered LDS filled by async global->LDS DMA; compute on buffer k
// overlaps the DMA of buffer k+1. Requires M%128==0, N%128==0, K%32==0.
// ---------------------------------------------------------------------------
#define BM 128
#define BN 128
#define BK 32
#define LDK 48   // 32 + 16 pad halves -> row stride 96 B (16-B aligned)

template<int EPI>
__global__ __launch_bounds__(256)
void gemm_bf16_wmma(int M, int N, int K,
                    const unsigned short* __restrict__ A, int lda, long long strA,
                    const unsigned short* __restrict__ B, int ldb, long long strB,
                    void* __restrict__ C, int ldc, long long strC,
                    const float* __restrict__ aux, int ldaux, long long strAux)
{
    __shared__ unsigned short As[2][BM * LDK];
    __shared__ unsigned short Bs[2][BM * LDK];

    const int bz  = blockIdx.z;
    A += (long long)bz * strA;
    B += (long long)bz * strB;

    const int tid  = threadIdx.x;
    const int lane = tid & 31;
    const int wave = tid >> 5;
    const int wm   = wave & 3;       // 4 waves along M
    const int wn   = wave >> 2;      // 2 waves along N
    const int half = lane >> 4;      // 0/1: K-half of the fragment
    const int l16  = lane & 15;

    const int m0 = blockIdx.y * BM;
    const int n0 = blockIdx.x * BN;

    v8f acc[2][4];
    const v8f vzero = {0.f, 0.f, 0.f, 0.f, 0.f, 0.f, 0.f, 0.f};
    for (int i = 0; i < 2; ++i)
        for (int j = 0; j < 4; ++j) acc[i][j] = vzero;

    // Staging map: each of 256 threads owns a 16-half (32 B) chunk of A and B.
    const int lr = tid >> 1;            // row 0..127 of the tile
    const int lc = (tid & 1) * 16;      // col 0 or 16 (halves)
    const unsigned short* aBase = A + (long long)(m0 + lr) * lda + lc;
    const unsigned short* bBase = B + (long long)(n0 + lr) * ldb + lc;

    const int nk = K / BK;

    // Prologue: DMA tile 0 into buffer 0.
    {
        async_load_b128(aBase,     &As[0][lr * LDK + lc]);
        async_load_b128(aBase + 8, &As[0][lr * LDK + lc + 8]);
        async_load_b128(bBase,     &Bs[0][lr * LDK + lc]);
        async_load_b128(bBase + 8, &Bs[0][lr * LDK + lc + 8]);
    }

    for (int kt = 0; kt < nk; ++kt) {
        const int cur = kt & 1;
        wait_async0();                 // my 4 DMAs for tile kt landed in LDS
        __syncthreads();               // everyone's tile kt present; buf cur^1 free

        if (kt + 1 < nk) {             // overlap: DMA tile kt+1 into other buffer
            const long long kb = (long long)(kt + 1) * BK;
            const unsigned short* ag = aBase + kb;
            const unsigned short* bg = bBase + kb;
            async_load_b128(ag,     &As[cur ^ 1][lr * LDK + lc]);
            async_load_b128(ag + 8, &As[cur ^ 1][lr * LDK + lc + 8]);
            async_load_b128(bg,     &Bs[cur ^ 1][lr * LDK + lc]);
            async_load_b128(bg + 8, &Bs[cur ^ 1][lr * LDK + lc + 8]);
            if (kt + 2 < nk) {         // pull tile kt+2 toward L2
                __builtin_prefetch((const void*)(ag + BK), 0, 1);
                __builtin_prefetch((const void*)(bg + BK), 0, 1);
            }
        }

        // A fragments: row m, elems 0..7 -> K=half*8.., elems 8..15 -> K=16+half*8..
        Frag16 af[2];
        for (int mt = 0; mt < 2; ++mt) {
            const unsigned short* p = &As[cur][(wm * 32 + mt * 16 + l16) * LDK + half * 8];
            af[mt].u[0] = *(const U4*)p;
            af[mt].u[1] = *(const U4*)(p + 16);
        }
        // B fragments: col n, elems 0..7 -> K=half*8.., elems 8..15 -> K=16+half*8..
        Frag16 bf[4];
        for (int nt = 0; nt < 4; ++nt) {
            const unsigned short* p = &Bs[cur][(wn * 64 + nt * 16 + l16) * LDK + half * 8];
            bf[nt].u[0] = *(const U4*)p;
            bf[nt].u[1] = *(const U4*)(p + 16);
        }
        for (int mt = 0; mt < 2; ++mt)
            for (int nt = 0; nt < 4; ++nt)
                acc[mt][nt] = __builtin_amdgcn_wmma_f32_16x16x32_bf16(
                    false, af[mt].v, false, bf[nt].v,
                    (short)0, acc[mt][nt], false, false);
    }

    float*          Cf = (float*)C          + (long long)bz * strC;
    unsigned short* Ch = (unsigned short*)C + (long long)bz * strC;
    const float*    Ax = aux ? aux + (long long)bz * strAux : nullptr;

    for (int mt = 0; mt < 2; ++mt)
        for (int nt = 0; nt < 4; ++nt)
            for (int r = 0; r < 8; ++r) {
                const int m = m0 + wm * 32 + mt * 16 + half * 8 + r;
                const int n = n0 + wn * 64 + nt * 16 + l16;
                const long long off = (long long)m * ldc + n;
                float v = acc[mt][nt][r];
                if constexpr (EPI == EPI_F32) {
                    Cf[off] = v;
                } else if constexpr (EPI == EPI_BF16) {
                    Ch[off] = f2bf(v);
                } else if constexpr (EPI == EPI_SILU_BF16) {
                    Ch[off] = f2bf(v / (1.f + __expf(-v)));
                } else if constexpr (EPI == EPI_SILU_F32) {
                    Cf[off] = v / (1.f + __expf(-v));
                } else if constexpr (EPI == EPI_HASH_BF16) {
                    float t = v * cGAMMA;
                    Ch[off] = f2bf(t / (1.f + fabsf(t)));
                } else if constexpr (EPI == EPI_MASK_F32) {
                    Cf[off] = v * cSCALE + ((n > m) ? -1e9f : 0.f);
                } else if constexpr (EPI == EPI_ADDRES_F32) {
                    Cf[off] = v + Ax[(long long)m * ldaux + n];
                } else if constexpr (EPI == EPI_MULAUX_BF16) {
                    Ch[off] = f2bf(v * Ax[(long long)m * ldaux + n]);
                }
            }
}

// ---------------------------------------------------------------------------
// RMSNorm (one block per row) -> bf16 output
// ---------------------------------------------------------------------------
__global__ __launch_bounds__(256)
void rmsnorm_bf16_kernel(const float* __restrict__ x, const float* __restrict__ w,
                         unsigned short* __restrict__ y, int H)
{
    const int row = blockIdx.x;
    const float* xr = x + (long long)row * H;
    __shared__ float sred[8];
    const int tid = threadIdx.x, lane = tid & 31, wv = tid >> 5;

    float ss = 0.f;
    const float4* x4 = (const float4*)xr;
    for (int i = tid; i < H / 4; i += 256) {
        float4 v = x4[i];
        ss += v.x * v.x + v.y * v.y + v.z * v.z + v.w * v.w;
    }
    for (int o = 16; o; o >>= 1) ss += __shfl_xor(ss, o, 32);
    if (lane == 0) sred[wv] = ss;
    __syncthreads();
    ss = 0.f;
    for (int i = 0; i < 8; ++i) ss += sred[i];
    const float rs = rsqrtf(ss / (float)H + cEPS);

    unsigned short* yr = y + (long long)row * H;
    for (int i = tid; i < H; i += 256)
        yr[i] = f2bf(xr[i] * rs * w[i]);
}

// ---------------------------------------------------------------------------
// RoPE: [S,H] f32 -> [NH,S,HD] bf16 (head-major for batched attention GEMMs)
// ---------------------------------------------------------------------------
__global__ __launch_bounds__(128)
void rope_kernel(const float* __restrict__ in, unsigned short* __restrict__ out)
{
    const int h = blockIdx.x, s = blockIdx.y, d = threadIdx.x;   // d in [0,128)
    const float* row = in + (long long)s * cH + h * cHD;
    const int dm = d & 63;
    const float invf = __powf(10000.f, -(float)dm * (1.f / 64.f));
    float c, sn;
    __sincosf((float)s * invf, &c, &sn);
    const float x  = row[d];
    const float xr = (d < 64) ? -row[d + 64] : row[d - 64];
    out[((long long)h * cS + s) * cHD + d] = f2bf(x * c + xr * sn);
}

// v: [S,H] f32  ->  vT: [NH,HD,S] bf16
__global__ __launch_bounds__(256)
void vtrans_kernel(const float* __restrict__ v, unsigned short* __restrict__ vT)
{
    const long long i = (long long)blockIdx.x * 256 + threadIdx.x;   // NH*HD*S
    const int h = (int)(i / (cHD * cS));
    const int r = (int)(i % (cHD * cS));
    const int d = r / cS;
    const int s = r % cS;
    vT[i] = f2bf(v[(long long)s * cH + h * cHD + d]);
}

// ---------------------------------------------------------------------------
// Row softmax over masked logits; writes true_attn f32 + bf16 copy for P*V
// ---------------------------------------------------------------------------
__global__ __launch_bounds__(256)
void softmax_kernel(const float* __restrict__ logits, float* __restrict__ pout,
                    unsigned short* __restrict__ pbf)
{
    const int row = blockIdx.x, h = blockIdx.y;
    const long long base = ((long long)h * cS + row) * cS;
    const float* lr = logits + base;
    __shared__ float sred[8];
    const int tid = threadIdx.x, lane = tid & 31, wv = tid >> 5;

    float mx = -1e30f;
    for (int i = tid; i < cS; i += 256) mx = fmaxf(mx, lr[i]);
    for (int o = 16; o; o >>= 1) mx = fmaxf(mx, __shfl_xor(mx, o, 32));
    if (lane == 0) sred[wv] = mx;
    __syncthreads();
    mx = sred[0];
    for (int i = 1; i < 8; ++i) mx = fmaxf(mx, sred[i]);
    __syncthreads();

    float sum = 0.f;
    for (int i = tid; i < cS; i += 256) sum += __expf(lr[i] - mx);
    for (int o = 16; o; o >>= 1) sum += __shfl_xor(sum, o, 32);
    if (lane == 0) sred[wv] = sum;
    __syncthreads();
    sum = 0.f;
    for (int i = 0; i < 8; ++i) sum += sred[i];
    const float inv = 1.f / sum;

    for (int i = tid; i < cS; i += 256) {
        const float p = __expf(lr[i] - mx) * inv;
        pout[base + i] = p;
        pbf[base + i]  = f2bf(p);
    }
}

// ---------------------------------------------------------------------------
// Transpose+convert: in[R,C] f32 -> out[C,R] bf16 (batched via blockIdx.z)
// ---------------------------------------------------------------------------
__global__ __launch_bounds__(256)
void transpose_bf16_kernel(const float* __restrict__ in, unsigned short* __restrict__ out,
                           int R, int C, long long sIn, long long sOut)
{
    __shared__ float tile[32][33];
    const int b  = blockIdx.z;
    const int c0 = blockIdx.x * 32, r0 = blockIdx.y * 32;
    const int tx = threadIdx.x & 31, ty = threadIdx.x >> 5;
    const float* ib = in + (long long)b * sIn;
    unsigned short* ob = out + (long long)b * sOut;
    for (int i = ty; i < 32; i += 8)
        tile[i][tx] = ib[(long long)(r0 + i) * C + c0 + tx];
    __syncthreads();
    for (int i = ty; i < 32; i += 8)
        ob[(long long)(c0 + i) * R + r0 + tx] = f2bf(tile[tx][i]);
}

// ---------------------------------------------------------------------------
// Host orchestration
// ---------------------------------------------------------------------------
extern "C" void kernel_launch(void* const* d_in, const int* in_sizes, int n_in,
                              void* d_out, int out_size, void* d_ws, size_t ws_size,
                              hipStream_t stream)
{
    (void)in_sizes; (void)n_in; (void)out_size; (void)ws_size;
    const float* hidden = (const float*)d_in[0];
    const float* wq     = (const float*)d_in[1];
    const float* wk     = (const float*)d_in[2];
    const float* wv     = (const float*)d_in[3];
    const float* wo     = (const float*)d_in[4];
    const float* rot1   = (const float*)d_in[5];
    const float* rot2   = (const float*)d_in[6];
    const float* ln1w   = (const float*)d_in[7];
    const float* ln2w   = (const float*)d_in[8];
    const float* wgate  = (const float*)d_in[9];
    const float* wup    = (const float*)d_in[10];
    const float* wdown  = (const float*)d_in[11];

    // bump allocator over d_ws
    char* wsp = (char*)d_ws;
    auto alloc = [&](size_t bytes) -> void* {
        void* p = wsp;
        wsp += (bytes + 255) & ~(size_t)255;
        return p;
    };
    typedef unsigned short bfh;
    const size_t SH  = (size_t)cS * cH;
    const size_t SS  = (size_t)cS * cS;
    const size_t SHD = (size_t)cS * cHD;

    bfh* wqT  = (bfh*)alloc((size_t)cH * cH * 2);
    bfh* wkT  = (bfh*)alloc((size_t)cH * cH * 2);
    bfh* wvT  = (bfh*)alloc((size_t)cH * cH * 2);
    bfh* woT  = (bfh*)alloc((size_t)cH * cH * 2);
    bfh* wgT  = (bfh*)alloc((size_t)cFF * cH * 2);
    bfh* wuT  = (bfh*)alloc((size_t)cFF * cH * 2);
    bfh* wdT  = (bfh*)alloc((size_t)cH * cFF * 2);
    bfh* r1T  = (bfh*)alloc((size_t)cNH * cHD * cHD * 2);
    bfh* r2T  = (bfh*)alloc((size_t)cNH * cHD * cHD * 2);
    bfh* xb   = (bfh*)alloc(SH * 2);
    float* qf = (float*)alloc(SH * 4);
    float* kf = (float*)alloc(SH * 4);
    float* vf = (float*)alloc(SH * 4);
    bfh* qrb  = (bfh*)alloc((size_t)cNH * SHD * 2);
    bfh* krb  = (bfh*)alloc((size_t)cNH * SHD * 2);
    bfh* vTb  = (bfh*)alloc((size_t)cNH * SHD * 2);
    bfh* T1   = (bfh*)alloc((size_t)cNH * SHD * 2);
    bfh* qh   = (bfh*)alloc((size_t)cNH * SHD * 2);
    bfh* kh   = (bfh*)alloc((size_t)cNH * SHD * 2);
    float* lg = (float*)alloc((size_t)cNH * SS * 4);
    bfh* Pb   = (bfh*)alloc((size_t)cNH * SS * 2);
    bfh* atc  = (bfh*)alloc(SH * 2);
    float* hb = (float*)alloc(SH * 4);
    bfh* hnb  = (bfh*)alloc(SH * 2);
    float* Gb = (float*)alloc((size_t)cS * cFF * 4);
    bfh* Mact = (bfh*)alloc((size_t)cS * cFF * 2);

    float* out0  = (float*)d_out;               // h + mlp  [S,H]
    float* draft = out0 + SH;                   // draft_attn [NH,S,S]
    float* trueA = draft + (size_t)cNH * SS;    // true_attn  [NH,S,S]

    const dim3 TB(256);
    // --- weight packing: f32 [K,N] -> bf16 [N,K] ---
    transpose_bf16_kernel<<<dim3(cH/32,  cH/32,  1),   TB, 0, stream>>>(wq,    wqT, cH,  cH,  0, 0);
    transpose_bf16_kernel<<<dim3(cH/32,  cH/32,  1),   TB, 0, stream>>>(wk,    wkT, cH,  cH,  0, 0);
    transpose_bf16_kernel<<<dim3(cH/32,  cH/32,  1),   TB, 0, stream>>>(wv,    wvT, cH,  cH,  0, 0);
    transpose_bf16_kernel<<<dim3(cH/32,  cH/32,  1),   TB, 0, stream>>>(wo,    woT, cH,  cH,  0, 0);
    transpose_bf16_kernel<<<dim3(cFF/32, cH/32,  1),   TB, 0, stream>>>(wgate, wgT, cH,  cFF, 0, 0);
    transpose_bf16_kernel<<<dim3(cFF/32, cH/32,  1),   TB, 0, stream>>>(wup,   wuT, cH,  cFF, 0, 0);
    transpose_bf16_kernel<<<dim3(cH/32,  cFF/32, 1),   TB, 0, stream>>>(wdown, wdT, cFF, cH,  0, 0);
    transpose_bf16_kernel<<<dim3(cHD/32, cHD/32, cNH), TB, 0, stream>>>(rot1,  r1T, cHD, cHD, cHD*cHD, cHD*cHD);
    transpose_bf16_kernel<<<dim3(cHD/32, cHD/32, cNH), TB, 0, stream>>>(rot2,  r2T, cHD, cHD, cHD*cHD, cHD*cHD);

    // --- attention input path ---
    rmsnorm_bf16_kernel<<<cS, TB, 0, stream>>>(hidden, ln1w, xb, cH);
    gemm_bf16_wmma<EPI_F32><<<dim3(cH/BN, cS/BM, 1), TB, 0, stream>>>(
        cS, cH, cH, xb, cH, 0, wqT, cH, 0, qf, cH, 0, nullptr, 0, 0);
    gemm_bf16_wmma<EPI_F32><<<dim3(cH/BN, cS/BM, 1), TB, 0, stream>>>(
        cS, cH, cH, xb, cH, 0, wkT, cH, 0, kf, cH, 0, nullptr, 0, 0);
    gemm_bf16_wmma<EPI_F32><<<dim3(cH/BN, cS/BM, 1), TB, 0, stream>>>(
        cS, cH, cH, xb, cH, 0, wvT, cH, 0, vf, cH, 0, nullptr, 0, 0);
    rope_kernel<<<dim3(cNH, cS), dim3(cHD), 0, stream>>>(qf, qrb);
    rope_kernel<<<dim3(cNH, cS), dim3(cHD), 0, stream>>>(kf, krb);
    vtrans_kernel<<<(unsigned)((size_t)cNH * SHD / 256), TB, 0, stream>>>(vf, vTb);

    // --- angle-LSH draft hash (batched per head) ---
    const dim3 gHash(cHD/BN, cS/BM, cNH);
    gemm_bf16_wmma<EPI_SILU_BF16><<<gHash, TB, 0, stream>>>(
        cS, cHD, cHD, qrb, cHD, (long long)SHD, r1T, cHD, (long long)cHD*cHD,
        T1, cHD, (long long)SHD, nullptr, 0, 0);
    gemm_bf16_wmma<EPI_HASH_BF16><<<gHash, TB, 0, stream>>>(
        cS, cHD, cHD, T1, cHD, (long long)SHD, r2T, cHD, (long long)cHD*cHD,
        qh, cHD, (long long)SHD, nullptr, 0, 0);
    gemm_bf16_wmma<EPI_SILU_BF16><<<gHash, TB, 0, stream>>>(
        cS, cHD, cHD, krb, cHD, (long long)SHD, r1T, cHD, (long long)cHD*cHD,
        T1, cHD, (long long)SHD, nullptr, 0, 0);
    gemm_bf16_wmma<EPI_HASH_BF16><<<gHash, TB, 0, stream>>>(
        cS, cHD, cHD, T1, cHD, (long long)SHD, r2T, cHD, (long long)cHD*cHD,
        kh, cHD, (long long)SHD, nullptr, 0, 0);

    // --- draft scores: qh @ kh^T  (Bt = kh directly) ---
    const dim3 gScore(cS/BN, cS/BM, cNH);
    gemm_bf16_wmma<EPI_F32><<<gScore, TB, 0, stream>>>(
        cS, cS, cHD, qh, cHD, (long long)SHD, kh, cHD, (long long)SHD,
        draft, cS, (long long)SS, nullptr, 0, 0);

    // --- true scores + causal mask, softmax, P@V ---
    gemm_bf16_wmma<EPI_MASK_F32><<<gScore, TB, 0, stream>>>(
        cS, cS, cHD, qrb, cHD, (long long)SHD, krb, cHD, (long long)SHD,
        lg, cS, (long long)SS, nullptr, 0, 0);
    softmax_kernel<<<dim3(cS, cNH), TB, 0, stream>>>(lg, trueA, Pb);
    // P [NH,S,S] @ V -> write into [S, NH*HD] (ldc=H, per-head offset HD)
    gemm_bf16_wmma<EPI_BF16><<<dim3(cHD/BN, cS/BM, cNH), TB, 0, stream>>>(
        cS, cHD, cS, Pb, cS, (long long)SS, vTb, cS, (long long)cHD*cS,
        atc, cH, (long long)cHD, nullptr, 0, 0);

    // --- output projection + residual ---
    gemm_bf16_wmma<EPI_ADDRES_F32><<<dim3(cH/BN, cS/BM, 1), TB, 0, stream>>>(
        cS, cH, cH, atc, cH, 0, woT, cH, 0, hb, cH, 0, hidden, cH, 0);

    // --- MLP ---
    rmsnorm_bf16_kernel<<<cS, TB, 0, stream>>>(hb, ln2w, hnb, cH);
    gemm_bf16_wmma<EPI_SILU_F32><<<dim3(cFF/BN, cS/BM, 1), TB, 0, stream>>>(
        cS, cFF, cH, hnb, cH, 0, wgT, cH, 0, Gb, cFF, 0, nullptr, 0, 0);
    gemm_bf16_wmma<EPI_MULAUX_BF16><<<dim3(cFF/BN, cS/BM, 1), TB, 0, stream>>>(
        cS, cFF, cH, hnb, cH, 0, wuT, cH, 0, Mact, cFF, 0, Gb, cFF, 0);
    gemm_bf16_wmma<EPI_ADDRES_F32><<<dim3(cH/BN, cS/BM, 1), TB, 0, stream>>>(
        cS, cH, cFF, Mact, cFF, 0, wdT, cFF, 0, out0, cH, 0, hb, cH, 0);
}
